// PointNet_SA_Module_2181843386698
// MI455X (gfx1250) — matrix operand order repaired and checked by reading, not compiled, not run on hardware
//
#include <hip/hip_runtime.h>
#include <hip/hip_bf16.h>

// ---------------- problem constants ----------------
#define BB_   16
#define NN_   4096
#define CPTS_ 64
#define MM_   1024
#define KK_   32
#define RAD2_ 0.04f          // 0.2^2
#define ROWS_ (BB_*MM_*KK_)  // 524288
#define EPS_  1e-5f

typedef __attribute__((ext_vector_type(16))) _Float16 v16h;
typedef __attribute__((ext_vector_type(8)))  _Float16 v8h;
typedef __attribute__((ext_vector_type(8)))  float    v8f;

// ---------------- 0. zero the BN stats accumulators ----------------
__global__ void zero_stats_kernel(float* s) { s[threadIdx.x] = 0.f; }  // <<<1,768>>>

// ---------------- 1. farthest point sampling (one block per batch) ----------------
__global__ void fps_kernel(const float* __restrict__ xyz, int* __restrict__ fps_inds) {
    const int b = blockIdx.x;
    __shared__ float mind[NN_];
    __shared__ float rv[32];
    __shared__ int   ri[32];
    __shared__ int   s_far;
    __shared__ float s_c[3];
    const int tid = threadIdx.x;                 // 1024 threads = 32 waves
    for (int p = tid; p < NN_; p += 1024) mind[p] = 1e10f;
    if (tid == 0) s_far = 0;
    __syncthreads();
    const float* xb = xyz + (size_t)b * NN_ * 3;
    for (int it = 0; it < MM_; ++it) {
        const int far = s_far;
        if (tid == 0) {
            fps_inds[b * MM_ + it] = far;
            s_c[0] = xb[far*3]; s_c[1] = xb[far*3+1]; s_c[2] = xb[far*3+2];
        }
        __syncthreads();
        const float cx = s_c[0], cy = s_c[1], cz = s_c[2];
        float bestv = -1.f; int besti = 0;
        for (int p = tid; p < NN_; p += 1024) {
            const float dx = xb[p*3] - cx, dy = xb[p*3+1] - cy, dz = xb[p*3+2] - cz;
            const float d  = dx*dx + dy*dy + dz*dz;
            const float md = fminf(mind[p], d);
            mind[p] = md;
            if (md > bestv || (md == bestv && p < besti)) { bestv = md; besti = p; }
        }
        #pragma unroll
        for (int off = 16; off > 0; off >>= 1) {            // wave32 reduction
            const float ov = __shfl_down(bestv, off, 32);
            const int   oi = __shfl_down(besti, off, 32);
            if (ov > bestv || (ov == bestv && oi < besti)) { bestv = ov; besti = oi; }
        }
        const int lane = tid & 31, wid = tid >> 5;
        if (lane == 0) { rv[wid] = bestv; ri[wid] = besti; }
        __syncthreads();
        if (wid == 0) {
            bestv = rv[lane]; besti = ri[lane];
            #pragma unroll
            for (int off = 16; off > 0; off >>= 1) {
                const float ov = __shfl_down(bestv, off, 32);
                const int   oi = __shfl_down(besti, off, 32);
                if (ov > bestv || (ov == bestv && oi < besti)) { bestv = ov; besti = oi; }
            }
            if (lane == 0) s_far = besti;
        }
        __syncthreads();
    }
}

// ---------------- 2. ball query: one thread per center, early-exit scan ----------------
__global__ void ball_query_kernel(const float* __restrict__ xyz,
                                  const int* __restrict__ fps_inds,
                                  float* __restrict__ out_newxyz,
                                  float* __restrict__ out_ginds_f,
                                  int*   __restrict__ g_inds_i) {
    const int t = blockIdx.x * blockDim.x + threadIdx.x;
    if (t >= BB_ * MM_) return;
    const int b = t >> 10;
    const int ci = fps_inds[t];
    const float* xb = xyz + (size_t)b * NN_ * 3;
    const float cx = xb[ci*3], cy = xb[ci*3+1], cz = xb[ci*3+2];
    out_newxyz[t*3+0] = cx; out_newxyz[t*3+1] = cy; out_newxyz[t*3+2] = cz;
    const long base = (long)t * KK_;
    int cnt = 0, first = NN_;
    for (int n = 0; n < NN_ && cnt < KK_; ++n) {
        const float dx = xb[n*3] - cx, dy = xb[n*3+1] - cy, dz = xb[n*3+2] - cz;
        if (dx*dx + dy*dy + dz*dz <= RAD2_) {
            if (cnt == 0) first = n;
            g_inds_i[base + cnt] = n;
            ++cnt;
        }
    }
    for (int j = cnt; j < KK_; ++j) g_inds_i[base + j] = first;
    for (int j = 0; j < KK_; ++j) out_ginds_f[base + j] = (float)g_inds_i[base + j];
}

// ---------------- 3. gather + concat -> f16 feats (stride 96, zero-padded 67..95) ----------------
__global__ void group_feats_kernel(const float* __restrict__ xyz,
                                   const float* __restrict__ points,
                                   const int*   __restrict__ g_inds_i,
                                   const float* __restrict__ newxyz,
                                   _Float16* __restrict__ feats) {
    const int row = blockIdx.x * blockDim.x + threadIdx.x;
    if (row >= ROWS_) return;
    const int bm = row >> 5;                 // b*M + m
    const int b  = bm >> 10;
    int idx = g_inds_i[row]; if (idx > NN_ - 1) idx = NN_ - 1;   // JAX OOB-gather clamp
    const float* p = xyz + ((size_t)b * NN_ + idx) * 3;
    const float* c = newxyz + (size_t)bm * 3;
    _Float16* f = feats + (size_t)row * 96;
    f[0] = (_Float16)(p[0] - c[0]);
    f[1] = (_Float16)(p[1] - c[1]);
    f[2] = (_Float16)(p[2] - c[2]);
    const float* pt = points + ((size_t)b * NN_ + idx) * CPTS_;
    #pragma unroll 8
    for (int j = 0; j < CPTS_; ++j) f[3 + j] = (_Float16)pt[j];
    #pragma unroll
    for (int j = 3 + CPTS_; j < 96; ++j) f[j] = (_Float16)0.f;
}

// ---------------- 4. pre-swizzle weights into WMMA B-fragment layout ----------------
// B (32x16, K x N) dense f16 layout: lane holds n = lane&15, halfs e: K = lg*8 + e (e<8), 16 + lg*8 + (e-8) (e>=8).
__global__ void prep_wfrag_kernel(const float* __restrict__ W, _Float16* __restrict__ Bf,
                                  int Cin, int NT) {
    const int nt = blockIdx.x % NT;
    const int kc = blockIdx.x / NT;
    const int lane = threadIdx.x;            // 32
    const int n  = lane & 15;
    const int lg = lane >> 4;
    _Float16* dst = Bf + ((size_t)(kc * NT + nt) * 32 + lane) * 16;
    #pragma unroll
    for (int e = 0; e < 16; ++e) {
        const int k = kc * 32 + (e < 8 ? lg * 8 + e : 16 + lg * 8 + (e - 8));
        dst[e] = (_Float16)((k < Cin) ? W[(nt * 16 + n) * Cin + k] : 0.f);
    }
}

// ---------------- 5. WMMA GEMM + fused BN-statistics reduction ----------------
// Weight fragments hoisted into VGPRs once per wave; each wave sweeps RG
// row-groups (amortizes B traffic 8x). __launch_bounds__(128,1) gives the
// allocator the full register file (worst case ~226 VGPRs, layer 2) so the
// fragment array stays in registers instead of spilling to scratch.
template<int KC, int NT, int STRIDE, int COUT, int RG>
__global__ void __launch_bounds__(128, 1)
gemm_wmma_kernel(const _Float16* __restrict__ A,
                 const _Float16* __restrict__ Bf,
                 float* __restrict__ H,
                 float* __restrict__ stats,   // [0..127] sum, [128..255] sumsq
                 int nrows) {
    __shared__ float lsum[COUT];
    __shared__ float lsq[COUT];
    const int tid = threadIdx.x;
    for (int c = tid; c < COUT; c += blockDim.x) { lsum[c] = 0.f; lsq[c] = 0.f; }
    __syncthreads();

    const int lane = tid & 31;
    const int wid  = tid >> 5;
    const int wv   = blockIdx.x * (blockDim.x >> 5) + wid;   // global wave id
    const int am   = lane & 15;
    const int lg   = lane >> 4;
    const int col0 = lane & 15;

    // hoist all B fragments into registers (wave-invariant)
    v16h bfr[KC * NT];
    #pragma unroll
    for (int f = 0; f < KC * NT; ++f)
        bfr[f] = *(const v16h*)(Bf + ((size_t)f * 32 + lane) * 16);

    float ssum[NT] = {};
    float ssq[NT]  = {};

    // grid is sized so every row-tile is valid: no guards -> simple CFG
    #pragma unroll 1
    for (int rt = 0; rt < RG; ++rt) {
        const int row0 = (wv * RG + rt) * 16;
        v8f acc[NT] = {};
        #pragma unroll
        for (int kc = 0; kc < KC; ++kc) {
            // A fragment: two contiguous 16B chunks per lane (ISA 7.12.2 layout)
            const _Float16* ap = A + (size_t)(row0 + am) * STRIDE + kc * 32 + lg * 8;
            const v8h alo = *(const v8h*)(ap);
            const v8h ahi = *(const v8h*)(ap + 16);
            v16h a;
            #pragma unroll
            for (int i = 0; i < 8; ++i) { a[i] = alo[i]; a[8 + i] = ahi[i]; }
            #pragma unroll
            for (int nt = 0; nt < NT; ++nt)
                acc[nt] = __builtin_amdgcn_wmma_f32_16x16x32_f16(
                    false, a, false, bfr[kc * NT + nt], (short)0, acc[nt], false, false);
        }
        // store h + register-level BN partials (each lane's 8 elems share a column)
        #pragma unroll
        for (int nt = 0; nt < NT; ++nt) {
            const int col = nt * 16 + col0;
            #pragma unroll
            for (int r = 0; r < 8; ++r) {
                const int row = row0 + r + lg * 8;
                const float v = acc[nt][r];
                H[(size_t)row * COUT + col] = v;
                ssum[nt] += v;
                ssq[nt]  += v * v;
            }
        }
    }
    // 2*NT LDS atomics per lane
    #pragma unroll
    for (int nt = 0; nt < NT; ++nt) {
        const int col = nt * 16 + col0;
        atomicAdd(&lsum[col], ssum[nt]);
        atomicAdd(&lsq[col],  ssq[nt]);
    }
    __syncthreads();
    for (int c = tid; c < COUT; c += blockDim.x) {
        atomicAdd(&stats[c],       lsum[c]);
        atomicAdd(&stats[128 + c], lsq[c]);
    }
}

// ---------------- 6. BN + ReLU -> f16 feats for next layer ----------------
__global__ void bn_relu_f16_kernel(const float* __restrict__ H,
                                   const float* __restrict__ stats,
                                   const float* __restrict__ g,
                                   const float* __restrict__ bv,
                                   _Float16* __restrict__ out,
                                   int cout, int nrows) {
    const long i = (long)blockIdx.x * blockDim.x + threadIdx.x;
    if (i >= (long)nrows * cout) return;
    const int  col = (int)(i % cout);
    const long row = i / cout;
    const float cnt  = (float)nrows;
    const float mean = stats[col] / cnt;
    const float var  = stats[128 + col] / cnt - mean * mean;
    const float sc   = g[col] * rsqrtf(var + EPS_);
    const float sh   = bv[col] - mean * sc;
    const float y    = fmaxf(sc * H[i] + sh, 0.f);
    out[row * cout + col] = (_Float16)y;
}

// ---------------- 7. final BN + ReLU + max over K ----------------
__global__ void bn_relu_max_kernel(const float* __restrict__ H,
                                   const float* __restrict__ stats,
                                   const float* __restrict__ g,
                                   const float* __restrict__ bv,
                                   float* __restrict__ out_np) {
    const int i = blockIdx.x * blockDim.x + threadIdx.x;   // over B*M*128
    if (i >= BB_ * MM_ * 128) return;
    const int col = i & 127;
    const int bm  = i >> 7;
    const float cnt  = (float)ROWS_;
    const float mean = stats[col] / cnt;
    const float var  = stats[128 + col] / cnt - mean * mean;
    const float sc   = g[col] * rsqrtf(var + EPS_);
    const float sh   = bv[col] - mean * sc;
    const float* hp  = H + (size_t)bm * KK_ * 128 + col;
    float mx = -1e30f;
    #pragma unroll 8
    for (int k = 0; k < KK_; ++k) {
        const float y = fmaxf(sc * hp[(size_t)k * 128] + sh, 0.f);
        mx = fmaxf(mx, y);
    }
    out_np[i] = mx;
}

// ---------------- host launcher ----------------
extern "C" void kernel_launch(void* const* d_in, const int* in_sizes, int n_in,
                              void* d_out, int out_size, void* d_ws, size_t ws_size,
                              hipStream_t stream) {
    (void)in_sizes; (void)n_in; (void)out_size; (void)ws_size;
    const float* xyz    = (const float*)d_in[0];
    const float* points = (const float*)d_in[1];
    const float* W0 = (const float*)d_in[2];
    const float* g0 = (const float*)d_in[3];
    const float* b0 = (const float*)d_in[4];
    const float* W1 = (const float*)d_in[5];
    const float* g1 = (const float*)d_in[6];
    const float* b1 = (const float*)d_in[7];
    const float* W2 = (const float*)d_in[8];
    const float* g2 = (const float*)d_in[9];
    const float* b2 = (const float*)d_in[10];

    float* out = (float*)d_out;
    float* out_newxyz = out;                                   // B*M*3   = 49152
    float* out_newpts = out + BB_ * MM_ * 3;                   // B*M*128 = 2097152
    float* out_ginds  = out + BB_ * MM_ * 3 + BB_ * MM_ * 128; // B*M*K   = 524288

    char* ws = (char*)d_ws;
    size_t off = 0;
    auto alloc = [&](size_t bytes) { size_t o = off; off = (off + bytes + 255) & ~(size_t)255; return o; };
    const size_t o_stats = alloc(3 * 256 * sizeof(float));
    const size_t o_fps   = alloc((size_t)BB_ * MM_ * sizeof(int));
    const size_t o_gind  = alloc((size_t)ROWS_ * sizeof(int));
    const size_t o_b0    = alloc((size_t)3 * 4 * 32 * 16 * sizeof(_Float16));
    const size_t o_b1    = alloc((size_t)2 * 4 * 32 * 16 * sizeof(_Float16));
    const size_t o_b2    = alloc((size_t)2 * 8 * 32 * 16 * sizeof(_Float16));
    const size_t o_feats = alloc((size_t)ROWS_ * 96 * sizeof(_Float16));
    const size_t o_h     = alloc((size_t)ROWS_ * 128 * sizeof(float));

    float*    stats   = (float*)(ws + o_stats);   // per-layer stride 256 floats
    int*      fpsinds = (int*)(ws + o_fps);
    int*      ginds   = (int*)(ws + o_gind);
    _Float16* bf0     = (_Float16*)(ws + o_b0);
    _Float16* bf1     = (_Float16*)(ws + o_b1);
    _Float16* bf2     = (_Float16*)(ws + o_b2);
    _Float16* feats   = (_Float16*)(ws + o_feats);
    float*    H       = (float*)(ws + o_h);

    zero_stats_kernel<<<1, 768, 0, stream>>>(stats);
    fps_kernel<<<BB_, 1024, 0, stream>>>(xyz, fpsinds);
    ball_query_kernel<<<(BB_ * MM_) / 256, 256, 0, stream>>>(xyz, fpsinds, out_newxyz, out_ginds, ginds);
    group_feats_kernel<<<ROWS_ / 256, 256, 0, stream>>>(xyz, points, ginds, out_newxyz, feats);

    prep_wfrag_kernel<<<3 * 4, 32, 0, stream>>>(W0, bf0, 67, 4);
    prep_wfrag_kernel<<<2 * 4, 32, 0, stream>>>(W1, bf1, 64, 4);
    prep_wfrag_kernel<<<2 * 8, 32, 0, stream>>>(W2, bf2, 64, 8);

    // 32768 row-groups, 8 per wave -> 4096 waves -> 1024 blocks of 4 waves
    const int RGPW    = 8;
    const int gemmblk = (ROWS_ / 16) / RGPW / 4;

    // layer 0: 96(padded from 67) -> 64
    gemm_wmma_kernel<3, 4, 96, 64, 8><<<gemmblk, 128, 0, stream>>>(feats, bf0, H, stats + 0 * 256, ROWS_);
    bn_relu_f16_kernel<<<(int)(((long)ROWS_ * 64 + 255) / 256), 256, 0, stream>>>(H, stats + 0 * 256, g0, b0, feats, 64, ROWS_);
    // layer 1: 64 -> 64
    gemm_wmma_kernel<2, 4, 64, 64, 8><<<gemmblk, 128, 0, stream>>>(feats, bf1, H, stats + 1 * 256, ROWS_);
    bn_relu_f16_kernel<<<(int)(((long)ROWS_ * 64 + 255) / 256), 256, 0, stream>>>(H, stats + 1 * 256, g1, b1, feats, 64, ROWS_);
    // layer 2: 64 -> 128
    gemm_wmma_kernel<2, 8, 64, 128, 8><<<gemmblk, 128, 0, stream>>>(feats, bf2, H, stats + 2 * 256, ROWS_);
    bn_relu_max_kernel<<<(BB_ * MM_ * 128) / 256, 256, 0, stream>>>(H, stats + 2 * 256, g2, b2, out_newpts);
}
